// FFMLP_5669356834525
// MI455X (gfx1250) — compile-verified, matrix operand order checked by reading, not run on hardware
//
#include <hip/hip_runtime.h>

typedef __attribute__((ext_vector_type(16))) _Float16 v16h;
typedef __attribute__((ext_vector_type(8)))  _Float16 v8h;
typedef __attribute__((ext_vector_type(2)))  _Float16 v2h;
typedef __attribute__((ext_vector_type(8)))  float    v8f;
typedef __attribute__((ext_vector_type(4)))  float    v4f;
typedef __attribute__((ext_vector_type(4)))  unsigned v4u;
typedef __attribute__((ext_vector_type(8)))  int      v8i;
typedef __attribute__((ext_vector_type(4)))  int      v4i;

#define INPUT_DIM 32
#define HIDDEN    128
#define NPARAMS   55296   // 128*32 + 3*128*128 + 16*128 (in halves)
#define W1_OFF    0
#define WH_OFF    4096    // hidden layer l (0..2) at WH_OFF + l*16384
#define WO_OFF    53248   // 4096 + 3*16384

#if defined(__gfx1250__) && __has_builtin(__builtin_amdgcn_tensor_load_to_lds) && \
    __has_builtin(__builtin_amdgcn_s_wait_tensorcnt)
#define USE_TDM 1
#else
#define USE_TDM 0
#endif

// ---- pre-pass: fp32 weights -> f16 in d_ws ----
__global__ __launch_bounds__(256) void cvt_weights_f16(
    const float* __restrict__ w, _Float16* __restrict__ o) {
  int i = blockIdx.x * 256 + threadIdx.x;
  if (i < NPARAMS) o[i] = (_Float16)w[i];
}

__device__ __forceinline__ float relu(float x) {
  // single v_med3_f32: clamp(x, 0, +inf) == max(x, 0) for non-NaN inputs
  return __builtin_amdgcn_fmed3f(x, 0.0f, __builtin_inff());
}

// Packed f32x2 -> f16x2 convert (v_cvt_pk_rtz_f16_f32), bit-cast to _Float16x2.
__device__ __forceinline__ v2h pk_f16(float a, float b) {
  auto p = __builtin_amdgcn_cvt_pkrtz(a, b);
  return __builtin_bit_cast(v2h, p);
}

// Load one A (weight) fragment 16x32 f16 from LDS.
// lane<16 : row h=16m+r, K halves {32kc+0..7, 32kc+16..23}
// lane>=16: row h=16m+r, K halves {32kc+8..15, 32kc+24..31}
__device__ __forceinline__ v16h load_a_frag(const _Float16* sW, int base, int K,
                                            int m, int kc, int r, int hi) {
  int h  = 16 * m + r;
  int k0 = 32 * kc + (hi ? 8 : 0);
  const _Float16* p = sW + base + h * K + k0;
  v8h lo = *(const v8h*)(p);        // ds_load_b128
  v8h up = *(const v8h*)(p + 16);   // ds_load_b128
  v16h a;
#pragma unroll
  for (int i = 0; i < 8; ++i) { a[i] = lo[i]; a[8 + i] = up[i]; }
  return a;
}

// Build next-layer B fragment from two C tiles (lane-local ReLU + packed cvt).
__device__ __forceinline__ v16h build_b_frag(const v8f c0, const v8f c1) {
  v16h b;
#pragma unroll
  for (int j = 0; j < 4; ++j) {
    v2h p0 = pk_f16(relu(c0[2 * j]), relu(c0[2 * j + 1]));
    v2h p1 = pk_f16(relu(c1[2 * j]), relu(c1[2 * j + 1]));
    b[2 * j]     = p0[0];
    b[2 * j + 1] = p0[1];
    b[8 + 2 * j]     = p1[0];
    b[8 + 2 * j + 1] = p1[1];
  }
  return b;
}

__global__ __launch_bounds__(256) void ffmlp_fused_fwd(
    const float* __restrict__ x, const _Float16* __restrict__ w16,
    float* __restrict__ out) {
  extern __shared__ _Float16 sW[];

#if USE_TDM
  // ---- Tensor Data Mover: one 1-D descriptor moves all 110592 B to LDS ----
  if (threadIdx.x == 0) {
    const unsigned long long ga = (unsigned long long)(uintptr_t)w16;
    const unsigned lds = (unsigned)(uintptr_t)sW;  // low 32b of flat = LDS offset
    const unsigned n8  = (NPARAMS * 2) / 8;        // 13824 8-byte elements

    v4u g0;
    g0[0] = 1u;                                    // count=1 (valid), user D#
    g0[1] = lds;                                   // lds_addr (bytes)
    g0[2] = (unsigned)(ga & 0xFFFFFFFFu);          // global_addr[31:0]
    g0[3] = (unsigned)((ga >> 32) & 0x01FFFFFFu)   // global_addr[56:32]
            | (2u << 30);                          // type=2 ("image")

    v8i g1;
    g1[0] = (int)(3u << 16);                       // wg_mask=0, data_size=8B
    g1[1] = (int)((n8 & 0xFFFFu) << 16);           // tensor_dim0[15:0]
    g1[2] = (int)((n8 >> 16) | (1u << 16));        // tensor_dim0[31:16] | tensor_dim1=1
    g1[3] = (int)(n8 << 16);                       // tensor_dim1[31:16]=0 | tile_dim0=n8
    g1[4] = 0;                                     // tile_dim1=0, tile_dim2=0 (1-D tile)
    g1[5] = (int)n8;                               // tensor_dim0_stride[31:0]
    g1[6] = 0;                                     // stride hi / dim1_stride lo
    g1[7] = 0;

    v4i z4 = {0, 0, 0, 0};
#if defined(__clang_major__) && __clang_major__ >= 23
    v8i z8 = {0, 0, 0, 0, 0, 0, 0, 0};
    __builtin_amdgcn_tensor_load_to_lds(g0, g1, z4, z4, z8, 0);
#else
    __builtin_amdgcn_tensor_load_to_lds(g0, g1, z4, z4, 0);
#endif
    __builtin_amdgcn_s_wait_tensorcnt(0);
  }
  __syncthreads();
#else
  // fallback: cooperative vector copy (55296 halves = 6912 uint4)
  {
    const uint4* src = (const uint4*)w16;
    uint4*       dst = (uint4*)sW;
    for (int i = threadIdx.x; i < NPARAMS / 8; i += 256) dst[i] = src[i];
  }
  __syncthreads();
#endif

  const int lane = threadIdx.x & 31;
  const int wave = threadIdx.x >> 5;
  const int r    = lane & 15;
  const int hi   = lane >> 4;
  const int rowBase = (blockIdx.x * 8 + wave) * 16;
  const int row     = rowBase + r;

  // ---- B0 = x^T tile (32 x 16): lane owns batch column r ----
  v16h b0;
  {
    const float* xr = x + (size_t)row * INPUT_DIM;
    const int c0 = hi ? 8 : 0;
    v4f f0 = __builtin_nontemporal_load((const v4f*)(xr + c0));
    v4f f1 = __builtin_nontemporal_load((const v4f*)(xr + c0 + 4));
    v4f f2 = __builtin_nontemporal_load((const v4f*)(xr + c0 + 16));
    v4f f3 = __builtin_nontemporal_load((const v4f*)(xr + c0 + 20));
#pragma unroll
    for (int j = 0; j < 2; ++j) {
      v2h p0 = pk_f16(f0[2 * j], f0[2 * j + 1]);
      v2h p1 = pk_f16(f1[2 * j], f1[2 * j + 1]);
      v2h p2 = pk_f16(f2[2 * j], f2[2 * j + 1]);
      v2h p3 = pk_f16(f3[2 * j], f3[2 * j + 1]);
      b0[2 * j] = p0[0];      b0[2 * j + 1] = p0[1];
      b0[4 + 2 * j] = p1[0];  b0[4 + 2 * j + 1] = p1[1];
      b0[8 + 2 * j] = p2[0];  b0[8 + 2 * j + 1] = p2[1];
      b0[12 + 2 * j] = p3[0]; b0[12 + 2 * j + 1] = p3[1];
    }
  }

  v8f c[8];  // y^T tiles: c[m] covers output rows h = 16m..16m+15

  // ---- layer 1: [128,32] * x^T, K=32; A-fragment loads pipelined 1 deep ----
  {
    v16h a = load_a_frag(sW, W1_OFF, 32, 0, 0, r, hi);
#pragma unroll
    for (int m = 0; m < 8; ++m) {
      v16h a_use = a;
      if (m < 7) a = load_a_frag(sW, W1_OFF, 32, m + 1, 0, r, hi);
      v8f z = {};
      c[m] = __builtin_amdgcn_wmma_f32_16x16x32_f16(false, a_use, false, b0,
                                                    (short)0, z, false, false);
    }
  }

  // ---- hidden layers 2..4: [128,128], K=128 = 4 chunks, pipelined ----
#pragma unroll
  for (int l = 0; l < 3; ++l) {
    const int base = WH_OFF + l * 16384;
    v16h B[4];
#pragma unroll
    for (int kc = 0; kc < 4; ++kc) B[kc] = build_b_frag(c[2 * kc], c[2 * kc + 1]);

    v16h a = load_a_frag(sW, base, HIDDEN, 0, 0, r, hi);
#pragma unroll
    for (int m = 0; m < 8; ++m) {
      v8f acc = {};
#pragma unroll
      for (int kc = 0; kc < 4; ++kc) {
        v16h a_use = a;
        if (!(m == 7 && kc == 3))
          a = load_a_frag(sW, base, HIDDEN, m + (kc == 3 ? 1 : 0), (kc + 1) & 3,
                          r, hi);
        acc = __builtin_amdgcn_wmma_f32_16x16x32_f16(false, a_use, false, B[kc],
                                                     (short)0, acc, false, false);
      }
      c[m] = acc;
    }
  }

  // ---- output layer: [16,128], K=128, no output activation ----
  {
    v16h B[4];
#pragma unroll
    for (int kc = 0; kc < 4; ++kc) B[kc] = build_b_frag(c[2 * kc], c[2 * kc + 1]);

    v16h a = load_a_frag(sW, WO_OFF, HIDDEN, 0, 0, r, hi);
    v8f acc = {};
#pragma unroll
    for (int kc = 0; kc < 4; ++kc) {
      v16h a_use = a;
      if (kc < 3) a = load_a_frag(sW, WO_OFF, HIDDEN, 0, kc + 1, r, hi);
      acc = __builtin_amdgcn_wmma_f32_16x16x32_f16(false, a_use, false, B[kc],
                                                   (short)0, acc, false, false);
    }
    // C tile = y^T (16 out x 16 batch): lane owns batch row rowBase+r,
    // VGPR v holds output element o = v + 8*hi -> 8 consecutive floats.
    float* op = out + (size_t)(rowBase + r) * 16 + (hi ? 8 : 0);
    v4f o0 = {acc[0], acc[1], acc[2], acc[3]};
    v4f o1 = {acc[4], acc[5], acc[6], acc[7]};
    __builtin_nontemporal_store(o0, (v4f*)op);
    __builtin_nontemporal_store(o1, (v4f*)(op + 4));
  }
}

extern "C" void kernel_launch(void* const* d_in, const int* in_sizes, int n_in,
                              void* d_out, int out_size, void* d_ws, size_t ws_size,
                              hipStream_t stream) {
  (void)n_in; (void)out_size; (void)ws_size;
  const float* x = (const float*)d_in[0];
  const float* w = (const float*)d_in[1];
  _Float16* w16  = (_Float16*)d_ws;

  cvt_weights_f16<<<(NPARAMS + 255) / 256, 256, 0, stream>>>(w, w16);

  const int rows   = in_sizes[0] / INPUT_DIM;   // 524288
  const int blocks = rows / 128;                // 8 waves * 16 rows per block

  const size_t shmem = (size_t)NPARAMS * sizeof(_Float16);  // 110592 B
  (void)hipFuncSetAttribute((const void*)ffmlp_fused_fwd,
                            hipFuncAttributeMaxDynamicSharedMemorySize,
                            (int)shmem);
  ffmlp_fused_fwd<<<blocks, 256, shmem, stream>>>(x, w16, (float*)d_out);
}
